// SP_CAM_Model_16415365005576
// MI455X (gfx1250) — compile-verified, matrix operand order checked by reading, not run on hardware
//
#include <hip/hip_runtime.h>
#include <hip/hip_bf16.h>

// ---------------------------------------------------------------------------
// Types for CDNA5 WMMA (wave32): A/B = 16 bf16 (8 VGPRs), C/D = 8 f32.
// ---------------------------------------------------------------------------
typedef __attribute__((ext_vector_type(16))) __bf16 v16bf;
typedef __attribute__((ext_vector_type(8)))  __bf16 v8bf;
typedef __attribute__((ext_vector_type(8)))  float  v8f;

static __device__ __forceinline__ unsigned short f2bfu(float f) {
  union { float f; unsigned u; } cv; cv.f = f;
  unsigned u = cv.u + 0x7FFFu + ((cv.u >> 16) & 1u);   // round-to-nearest-even
  return (unsigned short)(u >> 16);
}

// Fragment load: element e of lane l maps to K = (e<8 ? e : e+8) + (l>=16 ? 8:0).
// With row-major [row][K] bf16 data and the pointer pre-offset by kbase, the
// fragment is two contiguous 16-byte runs -> two global_load_b128.
static __device__ __forceinline__ v16bf ldfrag(const __bf16* __restrict__ p) {
  v8bf lo = *(const v8bf*)(p);        // K = c0+kbase .. +7
  v8bf hi = *(const v8bf*)(p + 16);   // K = c0+16+kbase .. +23
  return __builtin_shufflevector(lo, hi, 0, 1, 2, 3, 4, 5, 6, 7,
                                 8, 9, 10, 11, 12, 13, 14, 15);
}

static __device__ __forceinline__ v8f wmma_bf16(v16bf a, v16bf b, v8f c) {
  return __builtin_amdgcn_wmma_f32_16x16x32_bf16(false, a, false, b,
                                                 (short)0, c, false, false);
}

// ---------------------------------------------------------------------------
// Weight transform: w [CoutValid][Cin][ntaps(9|1)] f32  ->  Bt [CoutPad][Ktot]
// bf16 bits, Ktot = ntaps*Cin, K = t*Cin + c, rows n >= CoutValid zero-padded.
// ---------------------------------------------------------------------------
__global__ void wtrans(const float* __restrict__ w, unsigned short* __restrict__ Bt,
                       int Cin, int ntaps, int CoutValid, int CoutPad) {
  const size_t Ktot = (size_t)ntaps * Cin;
  const size_t total = (size_t)CoutPad * Ktot;
  size_t i = (size_t)blockIdx.x * blockDim.x + threadIdx.x;
  if (i >= total) return;
  const int n = (int)(i / Ktot);
  const size_t k = i % Ktot;
  const int c = (int)(k % Cin);
  const int t = (int)(k / Cin);
  Bt[i] = (n < CoutValid) ? f2bfu(w[((size_t)n * Cin + c) * ntaps + t])
                          : (unsigned short)0;
}

// ---------------------------------------------------------------------------
// im2col to bf16 on the B=8, 32x32 grid (3x3 pad-1 when ntaps==9, identity
// when ntaps==1). Zero padding baked in. Reads the channel-concat of
// inA (c < CsplitA, CA channels) and inB (remaining Cin-CsplitA channels),
// which fuses the x5||q concat for free.
//   A2[m][k], m = (b,y,x) flat, k = t*Cin + c, Ktot = ntaps*Cin
// ---------------------------------------------------------------------------
__global__ __launch_bounds__(256)
void im2col_bf16(const float* __restrict__ inA, const float* __restrict__ inB,
                 int CsplitA, unsigned short* __restrict__ A2, int Cin, int ntaps) {
  const size_t Ktot = (size_t)ntaps * Cin;
  const size_t total = (size_t)8192 * Ktot;
  size_t idx = (size_t)blockIdx.x * 256 + threadIdx.x;
  if (idx >= total) return;
  const size_t k = idx % Ktot;
  const int m = (int)(idx / Ktot);
  const int c = (int)(k % Cin);
  const int t = (int)(k / Cin);
  const int ky = (ntaps == 9) ? (t / 3) : 1;
  const int kx = (ntaps == 9) ? (t % 3) : 1;
  const int b = m >> 10;
  const int y = (m >> 5) & 31;
  const int x = m & 31;
  const int yy = y + ky - 1;
  const int xx = x + kx - 1;
  float v = 0.0f;
  if (yy >= 0 && yy < 32 && xx >= 0 && xx < 32) {
    v = (c < CsplitA)
        ? inA[(((size_t)b * CsplitA + c) * 32 + yy) * 32 + xx]
        : inB[(((size_t)b * (Cin - CsplitA) + (c - CsplitA)) * 32 + yy) * 32 + xx];
  }
  A2[idx] = f2bfu(v);
}

// ---------------------------------------------------------------------------
// bf16 WMMA GEMM: out[m][n] = A2[m][:] . Bt[n][:]  (+bias, optional ReLU),
// stored as [8][CoutStore][32][32]. Each wave computes a 32M x 32N macro-tile
// (2x2 WMMA tiles, 4 f32 accumulators). 4 waves/block -> 128 N per block.
// All fragment loads are pairs of global_load_b128; no VALU in the K loop
// beyond pointer bumps.
// ---------------------------------------------------------------------------
__global__ __launch_bounds__(128)
void gemm_wmma(const unsigned short* __restrict__ A2,  // [8192][Ktot]
               const unsigned short* __restrict__ Bt,  // [CoutPad][Ktot]
               const float* __restrict__ bias,         // [CoutStore] or null
               float* __restrict__ out,                // [8][CoutStore][32][32]
               int Ktot, int CoutPad, int CoutStore, int relu) {
  const int lane = threadIdx.x & 31;
  const int wave = threadIdx.x >> 5;
  const int n0 = blockIdx.y * 128 + wave * 32;
  if (n0 >= CoutPad) return;                 // no barriers in this kernel
  const int m0 = blockIdx.x * 32;            // two 16-row M tiles (same y row)
  const int nlane = lane & 15;
  const int kbase = (lane >> 4) << 3;        // +8 for upper half-wave

  const __bf16* a0p = (const __bf16*)A2 + (size_t)(m0 + nlane) * Ktot + kbase;
  const __bf16* a1p = a0p + (size_t)16 * Ktot;
  const __bf16* b0p = (const __bf16*)Bt + (size_t)(n0 + nlane) * Ktot + kbase;
  const __bf16* b1p = b0p + (size_t)16 * Ktot;

  v8f acc00 = {}, acc01 = {}, acc10 = {}, acc11 = {};
#pragma unroll 4
  for (int k = 0; k < Ktot; k += 32) {
    const v16bf a0 = ldfrag(a0p);
    const v16bf a1 = ldfrag(a1p);
    const v16bf b0 = ldfrag(b0p);
    const v16bf b1 = ldfrag(b1p);
    acc00 = wmma_bf16(a0, b0, acc00);
    acc01 = wmma_bf16(a0, b1, acc01);
    acc10 = wmma_bf16(a1, b0, acc10);
    acc11 = wmma_bf16(a1, b1, acc11);
    a0p += 32; a1p += 32; b0p += 32; b1p += 32;
  }

  // D layout: VGPR r -> M = r + (lane>=16 ? 8 : 0), N = lane&15
#pragma unroll
  for (int nt = 0; nt < 2; ++nt) {
    const int n = n0 + nt * 16 + nlane;
    if (n >= CoutStore) continue;
    const float bv = bias ? bias[n] : 0.0f;
#pragma unroll
    for (int mt = 0; mt < 2; ++mt) {
      const v8f acc = (nt == 0) ? (mt == 0 ? acc00 : acc10)
                                : (mt == 0 ? acc01 : acc11);
      const int p0 = m0 + mt * 16;
      const int b = p0 >> 10;
      const int y = (p0 >> 5) & 31;
      const int x0 = p0 & 31;
      float* op = out + (((size_t)b * CoutStore + n) * 32 + y) * 32 + x0 + kbase;
#pragma unroll
      for (int r = 0; r < 8; ++r) {
        float v = acc[r] + bv;
        if (relu) v = fmaxf(v, 0.0f);
        op[r] = v;
      }
    }
  }
}

// ---------------------------------------------------------------------------
// qf0: 3x3 conv, dilation 16, circular padding 16, 9 -> 32 channels, BN+ReLU.
// One thread per (b,y,x), all 32 output channels in registers; weights in LDS.
// ---------------------------------------------------------------------------
__global__ __launch_bounds__(256)
void qf0_conv(const float* __restrict__ probs,  // [8][9][512][512]
              const float* __restrict__ w,      // [32][9][3][3]
              const float* __restrict__ g, const float* __restrict__ bta,
              float* __restrict__ out) {        // [8][32][512][512]
  __shared__ float sw[81 * 32];                 // sw[(c*9+t)*32 + co]
  for (int i = threadIdx.x; i < 81 * 32; i += 256) {
    int co = i & 31, ct = i >> 5;
    sw[i] = w[((size_t)co * 9 + (ct / 9)) * 9 + (ct % 9)];
  }
  __syncthreads();

  size_t idx = (size_t)blockIdx.x * 256 + threadIdx.x;   // 8*512*512 threads
  if (idx >= (size_t)8 * 512 * 512) return;
  const int x = (int)(idx & 511);
  const int y = (int)((idx >> 9) & 511);
  const int b = (int)(idx >> 18);

  float acc[32];
#pragma unroll
  for (int co = 0; co < 32; ++co) acc[co] = 0.0f;

  for (int c = 0; c < 9; ++c) {
    const float* ip = probs + ((size_t)b * 9 + c) * 512 * 512;
#pragma unroll
    for (int ky = 0; ky < 3; ++ky) {
      const int yy = (y + 16 * ky - 16 + 512) & 511;     // circular
#pragma unroll
      for (int kx = 0; kx < 3; ++kx) {
        const int xx = (x + 16 * kx - 16 + 512) & 511;
        const float v = ip[(size_t)yy * 512 + xx];
        const float* swp = &sw[((c * 3 + ky) * 3 + kx) * 32];
#pragma unroll
        for (int co = 0; co < 32; ++co) acc[co] += v * swp[co];
      }
    }
  }
  float* op = out + ((size_t)b * 32 * 512 + y) * 512 + x;
#pragma unroll
  for (int co = 0; co < 32; ++co)
    op[(size_t)co * 262144] = fmaxf(acc[co] * g[co] + bta[co], 0.0f);
}

// ---------------------------------------------------------------------------
// Generic direct 3x3 stride-2 pad-1 conv with BN+ReLU (qf1..qf4, small K).
// ---------------------------------------------------------------------------
__global__ __launch_bounds__(256)
void conv3x3_s2(const float* __restrict__ in, const float* __restrict__ w,
                const float* __restrict__ g, const float* __restrict__ bta,
                float* __restrict__ out, int Cin, int Cout, int Hin, int Win) {
  const int Hout = Hin >> 1, Wout = Win >> 1;
  const size_t total = (size_t)8 * Cout * Hout * Wout;
  size_t idx = (size_t)blockIdx.x * 256 + threadIdx.x;
  if (idx >= total) return;
  const int x = (int)(idx % Wout);
  size_t r = idx / Wout;
  const int y = (int)(r % Hout); r /= Hout;
  const int co = (int)(r % Cout);
  const int b  = (int)(r / Cout);

  float acc = 0.0f;
  const float* wp = w + (size_t)co * Cin * 9;
  for (int c = 0; c < Cin; ++c) {
    const float* ip = in + ((size_t)b * Cin + c) * Hin * Win;
#pragma unroll
    for (int ky = 0; ky < 3; ++ky) {
      const int yy = 2 * y + ky - 1;
      if (yy < 0 || yy >= Hin) continue;
#pragma unroll
      for (int kx = 0; kx < 3; ++kx) {
        const int xx = 2 * x + kx - 1;
        if (xx < 0 || xx >= Win) continue;
        acc += ip[(size_t)yy * Win + xx] * wp[(c * 3 + ky) * 3 + kx];
      }
    }
  }
  out[idx] = fmaxf(acc * g[co] + bta[co], 0.0f);
}

// ---------------------------------------------------------------------------
// Per-half softmax over 9 channels of y3 [8][18][32][32].
// ---------------------------------------------------------------------------
__global__ __launch_bounds__(256)
void softmax9x2(const float* __restrict__ y3, float* __restrict__ sm) {
  const size_t total = (size_t)8 * 1024 * 2;
  size_t idx = (size_t)blockIdx.x * 256 + threadIdx.x;
  if (idx >= total) return;
  const int half = (int)(idx & 1);
  size_t r = idx >> 1;
  const int pos = (int)(r & 1023);
  const int b   = (int)(r >> 10);
  const float* p = y3 + ((size_t)b * 18 + half * 9) * 1024 + pos;
  float mx = -3.0e38f;
#pragma unroll
  for (int k = 0; k < 9; ++k) mx = fmaxf(mx, p[(size_t)k * 1024]);
  float e[9], s = 0.0f;
#pragma unroll
  for (int k = 0; k < 9; ++k) { e[k] = __expf(p[(size_t)k * 1024] - mx); s += e[k]; }
  const float inv = 1.0f / s;
  float* o = sm + ((size_t)b * 18 + half * 9) * 1024 + pos;
#pragma unroll
  for (int k = 0; k < 9; ++k) o[(size_t)k * 1024] = e[k] * inv;
}

// ---------------------------------------------------------------------------
// Fused noliner + upfeat: out[b,co,y,x] = sum_{i,j} sm[b,off+3i+j,y+i-1,x+j-1]
//                                          * logits[b,co,y+i-1,x+j-1]
// (off = 0 for co==0 (bg), 9 for co>=1 (fg)).
// ---------------------------------------------------------------------------
__global__ __launch_bounds__(256)
void sp_cam_out(const float* __restrict__ logits,  // [8][21][32][32]
                const float* __restrict__ sm,      // [8][18][32][32]
                float* __restrict__ out) {         // [8][21][32][32]
  const size_t total = (size_t)8 * 21 * 1024;
  size_t idx = (size_t)blockIdx.x * 256 + threadIdx.x;
  if (idx >= total) return;
  const int x = (int)(idx & 31);
  const int y = (int)((idx >> 5) & 31);
  size_t r = idx >> 10;
  const int co = (int)(r % 21);
  const int b  = (int)(r / 21);
  const int off = (co == 0) ? 0 : 9;
  float acc = 0.0f;
#pragma unroll
  for (int i = 0; i < 3; ++i) {
    const int yy = y + i - 1;
    if (yy < 0 || yy >= 32) continue;
#pragma unroll
    for (int j = 0; j < 3; ++j) {
      const int xx = x + j - 1;
      if (xx < 0 || xx >= 32) continue;
      const float pr = sm[(((size_t)b * 18 + off + 3 * i + j) * 32 + yy) * 32 + xx];
      const float lv = logits[(((size_t)b * 21 + co) * 32 + yy) * 32 + xx];
      acc += pr * lv;
    }
  }
  out[idx] = acc;
}

// ---------------------------------------------------------------------------
// Host launcher
// ---------------------------------------------------------------------------
static inline unsigned cdiv(size_t a, size_t b) { return (unsigned)((a + b - 1) / b); }

extern "C" void kernel_launch(void* const* d_in, const int* in_sizes, int n_in,
                              void* d_out, int out_size, void* d_ws, size_t ws_size,
                              hipStream_t stream) {
  (void)in_sizes; (void)n_in; (void)out_size; (void)ws_size;
  const float* probs = (const float*)d_in[0];
  const float* x5    = (const float*)d_in[1];
  const float* qf0_w = (const float*)d_in[2];
  const float* qf0_g = (const float*)d_in[3];
  const float* qf0_b = (const float*)d_in[4];
  const float* qf1_w = (const float*)d_in[5];
  const float* qf1_g = (const float*)d_in[6];
  const float* qf1_b = (const float*)d_in[7];
  const float* qf2_w = (const float*)d_in[8];
  const float* qf2_g = (const float*)d_in[9];
  const float* qf2_b = (const float*)d_in[10];
  const float* qf3_w = (const float*)d_in[11];
  const float* qf3_g = (const float*)d_in[12];
  const float* qf3_b = (const float*)d_in[13];
  const float* qf4_w = (const float*)d_in[14];
  const float* qf4_g = (const float*)d_in[15];
  const float* qf4_b = (const float*)d_in[16];
  const float* tp0_w = (const float*)d_in[17];
  const float* tp0_b = (const float*)d_in[18];
  const float* tp1_w = (const float*)d_in[19];
  const float* tp1_b = (const float*)d_in[20];
  const float* tp2_w = (const float*)d_in[21];
  const float* tp2_b = (const float*)d_in[22];
  const float* tp3_w = (const float*)d_in[23];
  const float* tp3_b = (const float*)d_in[24];
  const float* cls_w = (const float*)d_in[25];

  char* ws = (char*)d_ws;
  size_t off = 0;
  auto alloc = [&](size_t bytes) { size_t o = off; off += (bytes + 255) & ~(size_t)255; return o; };

  unsigned short* Bt0 = (unsigned short*)(ws + alloc((size_t)1024 * 19584 * 2));
  unsigned short* Bt1 = (unsigned short*)(ws + alloc((size_t)256 * 9216 * 2));
  unsigned short* Bt2 = (unsigned short*)(ws + alloc((size_t)128 * 2304 * 2));
  unsigned short* Bt3 = (unsigned short*)(ws + alloc((size_t)32 * 128 * 2));
  unsigned short* Btc = (unsigned short*)(ws + alloc((size_t)32 * 2048 * 2));
  float* logits = (float*)(ws + alloc((size_t)8 * 21 * 1024 * 4));
  float* x2     = (float*)(ws + alloc((size_t)8 * 32 * 256 * 256 * 4));
  float* x3     = (float*)(ws + alloc((size_t)8 * 64 * 128 * 128 * 4));
  float* x4     = (float*)(ws + alloc((size_t)8 * 128 * 64 * 64 * 4));
  float* qb     = (float*)(ws + alloc((size_t)8 * 128 * 1024 * 4));

  // Region R: x1 (268 MB qf0 output) is consumed by qf1 before the A2 im2col
  // buffer (321 MB) aliases the same base; y* live after A2.
  size_t Rbase = off;
  float* x1 = (float*)(ws + Rbase);              // 8*32*512*512*4 bytes
  size_t r = Rbase;
  auto ralloc = [&](size_t bytes) { size_t o = r; r += (bytes + 255) & ~(size_t)255; return o; };
  unsigned short* A2 = (unsigned short*)(ws + ralloc((size_t)8192 * 19584 * 2));
  float* y0  = (float*)(ws + ralloc((size_t)8 * 1024 * 1024 * 4));
  float* y1  = (float*)(ws + ralloc((size_t)8 * 256 * 1024 * 4));
  float* y2  = (float*)(ws + ralloc((size_t)8 * 128 * 1024 * 4));
  float* y3  = (float*)(ws + ralloc((size_t)8 * 18 * 1024 * 4));
  float* smb = (float*)(ws + ralloc((size_t)8 * 18 * 1024 * 4));

  // --- weight transforms (bf16 [CoutPad][Ktot]) ---
  wtrans<<<cdiv((size_t)1024 * 19584, 256), 256, 0, stream>>>(tp0_w, Bt0, 2176, 9, 1024, 1024);
  wtrans<<<cdiv((size_t)256 * 9216, 256), 256, 0, stream>>>(tp1_w, Bt1, 1024, 9, 256, 256);
  wtrans<<<cdiv((size_t)128 * 2304, 256), 256, 0, stream>>>(tp2_w, Bt2, 256, 9, 128, 128);
  wtrans<<<cdiv((size_t)32 * 128, 256), 256, 0, stream>>>(tp3_w, Bt3, 128, 1, 18, 32);
  wtrans<<<cdiv((size_t)32 * 2048, 256), 256, 0, stream>>>(cls_w, Btc, 2048, 1, 21, 32);

  // --- get_qfeats chain (runs first so x1 is dead before A2 aliases it) ---
  qf0_conv<<<8192, 256, 0, stream>>>(probs, qf0_w, qf0_g, qf0_b, x1);
  conv3x3_s2<<<cdiv((size_t)8 * 32 * 256 * 256, 256), 256, 0, stream>>>(x1, qf1_w, qf1_g, qf1_b, x2, 32, 32, 512, 512);
  conv3x3_s2<<<cdiv((size_t)8 * 64 * 128 * 128, 256), 256, 0, stream>>>(x2, qf2_w, qf2_g, qf2_b, x3, 32, 64, 256, 256);
  conv3x3_s2<<<cdiv((size_t)8 * 128 * 64 * 64, 256), 256, 0, stream>>>(x3, qf3_w, qf3_g, qf3_b, x4, 64, 128, 128, 128);
  conv3x3_s2<<<cdiv((size_t)8 * 128 * 32 * 32, 256), 256, 0, stream>>>(x4, qf4_w, qf4_g, qf4_b, qb, 128, 128, 64, 64);

  // --- classifier: logits = conv1x1(x5, cls_w), N padded to 32 ---
  im2col_bf16<<<cdiv((size_t)8192 * 2048, 256), 256, 0, stream>>>(x5, x5, 2048, A2, 2048, 1);
  gemm_wmma<<<dim3(256, 1), 128, 0, stream>>>(A2, Btc, nullptr, logits, 2048, 32, 21, 0);

  // --- tran-conv-parms chain (im2col fuses concat [x5 || q]) ---
  im2col_bf16<<<cdiv((size_t)8192 * 19584, 256), 256, 0, stream>>>(x5, qb, 2048, A2, 2176, 9);
  gemm_wmma<<<dim3(256, 8), 128, 0, stream>>>(A2, Bt0, tp0_b, y0, 19584, 1024, 1024, 1);

  im2col_bf16<<<cdiv((size_t)8192 * 9216, 256), 256, 0, stream>>>(y0, y0, 1024, A2, 1024, 9);
  gemm_wmma<<<dim3(256, 2), 128, 0, stream>>>(A2, Bt1, tp1_b, y1, 9216, 256, 256, 1);

  im2col_bf16<<<cdiv((size_t)8192 * 2304, 256), 256, 0, stream>>>(y1, y1, 256, A2, 256, 9);
  gemm_wmma<<<dim3(256, 1), 128, 0, stream>>>(A2, Bt2, tp2_b, y2, 2304, 128, 128, 1);

  im2col_bf16<<<cdiv((size_t)8192 * 128, 256), 256, 0, stream>>>(y2, y2, 128, A2, 128, 1);
  gemm_wmma<<<dim3(256, 1), 128, 0, stream>>>(A2, Bt3, tp3_b, y3, 128, 32, 18, 1);

  // --- DRM + sp_cam ---
  softmax9x2<<<cdiv((size_t)8 * 1024 * 2, 256), 256, 0, stream>>>(y3, smb);
  sp_cam_out<<<cdiv((size_t)8 * 21 * 1024, 256), 256, 0, stream>>>(logits, smb, (float*)d_out);
}
